// STGCN_16466904613202
// MI455X (gfx1250) — matrix operand order repaired and checked by reading, not compiled, not run on hardware
//
#include <hip/hip_runtime.h>
#include <hip/hip_bf16.h>

// ============================================================================
// ST-GCN forward for MI455X (gfx1250, wave32, WMMA).
//
// Roofline: ~56 GFLOP total, dominated by an M=8192,K=3072,N=1024 contraction.
// Intermediates (~113 MB) fit in the 192 MB L2, so the kernel is matrix-
// throughput bound -> use v_wmma_f32_16x16x32_f16 with f32 accumulation.
// BN is folded to per-channel scale/shift, tcn biases folded into BN shift,
// ReLU + f16 quantize fused into the GEMM epilogue.
//
// GEMM tiling: 128x64 block tile, 8 waves, 32x32 wave tile -> each K-step is
// 16 dword A + 32 f16 B loads per lane feeding 4 WMMAs, halving L2 traffic vs a
// 16x32 wave tile and giving the scheduler 4 independent accumulator chains.
//
// Workspace required: ~113 MB.
// ============================================================================

#define EPS 1e-5f

typedef __attribute__((ext_vector_type(16))) _Float16 v16h;
typedef __attribute__((ext_vector_type(8)))  float    v8f;

// -------- input index map (setup_inputs() dict insertion order, flattened) --
enum {
  IN_JOINTS = 0, IN_A = 1,
  PE_W1 = 2, PE_B1, PE_BNG, PE_BNB, PE_BNM, PE_BNV, PE_W2, PE_B2,   // pos_embed
  SK_W1 = 10, SK_B1, SK_BNG, SK_BNB, SK_BNM, SK_BNV, SK_W2, SK_B2,  // sk_feat
  B1_GCNW = 18, B1_GCNB, B1_IMP, B1_TCNW, B1_TCNB,
  B1_BN1G, B1_BN1B, B1_BN1M, B1_BN1V, B1_BN2G, B1_BN2B, B1_BN2M, B1_BN2V,
  B2_GCNW = 31, B2_GCNB, B2_IMP, B2_TCNW, B2_TCNB,
  B2_BN1G, B2_BN1B, B2_BN1M, B2_BN1V, B2_BN2G, B2_BN2B, B2_BN2M, B2_BN2V,
  B2_RESW = 44, B2_RESB, B2_RBG, B2_RBB, B2_RBM, B2_RBV,
  CJ_W = 50, CJ_B = 51
};

// prep block layout (floats):
// [0:64)  sk bn scale   [64:128) sk bn shift
// [128:192) block1 bn1 scale  [192:256) block1 bn1 shift
// [256:320) block1 bn2 scale  [320:384) block1 bn2 shift (tcn_b folded)
// [384:386) block2 bn1 scale  [386:388) block2 bn1 shift
// [388:390) pos constant (2)

struct PrepArgs {
  const float *sk_g, *sk_b, *sk_m, *sk_v;
  const float *b1g, *b1b, *b1m, *b1v;
  const float *b2g, *b2b, *b2m, *b2v, *tcnb1;
  const float *k2g, *k2b, *k2m, *k2v;
  const float *pe_b1, *pe_g, *pe_b, *pe_m, *pe_v, *pe_w2, *pe_b2;
};

__global__ void k_prep(PrepArgs a, float* prep) {
  int j = threadIdx.x;                       // 64 threads
  if (j >= 64) return;
  float ss = a.sk_g[j] * rsqrtf(a.sk_v[j] + EPS);
  prep[j] = ss;
  prep[64 + j] = a.sk_b[j] - a.sk_m[j] * ss;
  float s1 = a.b1g[j] * rsqrtf(a.b1v[j] + EPS);
  prep[128 + j] = s1;
  prep[192 + j] = a.b1b[j] - a.b1m[j] * s1;
  float s2 = a.b2g[j] * rsqrtf(a.b2v[j] + EPS);
  prep[256 + j] = s2;
  prep[320 + j] = a.b2b[j] + (a.tcnb1[j] - a.b2m[j]) * s2;
  if (j < 2) {
    float s3 = a.k2g[j] * rsqrtf(a.k2v[j] + EPS);
    prep[384 + j] = s3;
    prep[386 + j] = a.k2b[j] - a.k2m[j] * s3;
    // pos_embed on a zero input: h = b1 -> BN -> relu -> @ w2.T + b2
    float acc = a.pe_b2[j];
    for (int t = 0; t < 64; ++t) {
      float h = a.pe_b1[t];
      float sp = a.pe_g[t] * rsqrtf(a.pe_v[t] + EPS);
      h = (h - a.pe_m[t]) * sp + a.pe_b[t];
      h = fmaxf(h, 0.f);
      acc += h * a.pe_w2[j * 64 + t];
    }
    prep[388 + j] = acc;
  }
}

// sk_feat MLP: xr = joints - joints[:,0]; 3->64 BN relu -> 2; + pos const
__global__ void k_feat(const float* __restrict__ joints,
                       const float* __restrict__ w1, const float* __restrict__ b1,
                       const float* __restrict__ w2, const float* __restrict__ b2,
                       const float* __restrict__ prep, float* __restrict__ f) {
  int i = blockIdx.x * blockDim.x + threadIdx.x;   // b*1024 + v
  if (i >= 128 * 1024) return;
  int b = i >> 10, v = i & 1023;
  const float* p  = joints + ((size_t)b * 1024 + v) * 3;
  const float* p0 = joints + (size_t)b * 1024 * 3;
  float x0 = p[0] - p0[0], x1 = p[1] - p0[1], x2 = p[2] - p0[2];
  float o0 = b2[0] + prep[388], o1 = b2[1] + prep[389];
  #pragma unroll 8
  for (int j = 0; j < 64; ++j) {
    float h = w1[j * 3] * x0 + w1[j * 3 + 1] * x1 + w1[j * 3 + 2] * x2 + b1[j];
    h = fmaxf(h * prep[j] + prep[64 + j], 0.f);
    o0 += h * w2[j];
    o1 += h * w2[64 + j];
  }
  f[(size_t)i * 2]     = o0;
  f[(size_t)i * 2 + 1] = o1;
}

// block1 gcn 1x1 (2 -> 192 ch), laid out for the big GEMM:
// H[row=b*64+c][col=k*1024+v] = f0*W[k64c][0] + f1*W[k64c][1] + bias
__global__ void k_gcn1(const float* __restrict__ f, const float* __restrict__ gw,
                       const float* __restrict__ gb, _Float16* __restrict__ H) {
  long long i = (long long)blockIdx.x * blockDim.x + threadIdx.x;
  if (i >= (long long)8192 * 3072) return;
  int col = (int)(i % 3072), row = (int)(i / 3072);
  int b = row >> 6, c = row & 63;
  int k = col >> 10, v = col & 1023;
  int oc = k * 64 + c;
  const float* fp = f + ((size_t)b * 1024 + v) * 2;
  H[i] = (_Float16)(fp[0] * gw[oc * 2] + fp[1] * gw[oc * 2 + 1] + gb[oc]);
}

// block2 gcn 1x1 (64 -> 6 ch): H2[row=b*2+c2][col=k*1024+v]
__global__ void k_gcn2(const _Float16* __restrict__ X1, const float* __restrict__ gw2,
                       const float* __restrict__ gb2, _Float16* __restrict__ H2) {
  int i = blockIdx.x * blockDim.x + threadIdx.x;   // 128*6*1024
  if (i >= 128 * 6 * 1024) return;
  int v = i & 1023, t = i >> 10;
  int kc = t % 6, b = t / 6;
  int k = kc >> 1, c2 = kc & 1;
  float acc = gb2[kc];
  const _Float16* xb = X1 + (size_t)b * 64 * 1024 + v;
  #pragma unroll 16
  for (int ci = 0; ci < 64; ++ci) acc += (float)xb[(size_t)ci * 1024] * gw2[kc * 64 + ci];
  H2[((size_t)(b * 2 + c2)) * 3072 + (size_t)k * 1024 + v] = (_Float16)acc;
}

// block2 tcn(center tap) + bn2 + residual(1x1 conv + bn) + relu, store transposed
// X2T[c2*1024+v][b] for the final GEMM's B operand.
__global__ void k_tcn2res(const _Float16* __restrict__ Y2, const _Float16* __restrict__ X1,
                          const float* __restrict__ tw2, const float* __restrict__ tb2,
                          const float* __restrict__ g2, const float* __restrict__ bb2,
                          const float* __restrict__ m2, const float* __restrict__ v2,
                          const float* __restrict__ rw, const float* __restrict__ rb,
                          const float* __restrict__ rbg, const float* __restrict__ rbb,
                          const float* __restrict__ rbm, const float* __restrict__ rbv,
                          _Float16* __restrict__ X2T) {
  int i = blockIdx.x * blockDim.x + threadIdx.x;   // 128*2*1024
  if (i >= 128 * 2 * 1024) return;
  int v = i & 1023, t = i >> 10;
  int c2 = t & 1, b = t >> 1;
  float z = tb2[c2];
  z += (float)Y2[((size_t)b * 2 + 0) * 1024 + v] * tw2[c2 * 6 + 0 * 3 + 1];
  z += (float)Y2[((size_t)b * 2 + 1) * 1024 + v] * tw2[c2 * 6 + 1 * 3 + 1];
  float s2 = g2[c2] * rsqrtf(v2[c2] + EPS);
  z = (z - m2[c2]) * s2 + bb2[c2];
  float r = rb[c2];
  const _Float16* xb = X1 + (size_t)b * 64 * 1024 + v;
  #pragma unroll 16
  for (int ci = 0; ci < 64; ++ci) r += (float)xb[(size_t)ci * 1024] * rw[c2 * 64 + ci];
  float sr = rbg[c2] * rsqrtf(rbv[c2] + EPS);
  r = (r - rbm[c2]) * sr + rbb[c2];
  X2T[((size_t)(c2 * 1024 + v)) * 128 + b] = (_Float16)fmaxf(z + r, 0.f);
}

// generic f32 -> f16 convert with optional elementwise multiplier (A * imp)
__global__ void k_cvt(const float* __restrict__ a, const float* __restrict__ mul,
                      _Float16* __restrict__ o, long long n) {
  long long i = (long long)blockIdx.x * blockDim.x + threadIdx.x;
  if (i < n) o[i] = (_Float16)(mul ? a[i] * mul[i] : a[i]);
}

// tcn1 center-tap weights: W1[co][ci] = tcn_w[co][ci][1][0]
__global__ void k_w1(const float* __restrict__ tw, _Float16* __restrict__ o) {
  int i = blockIdx.x * blockDim.x + threadIdx.x;
  if (i >= 64 * 64) return;
  int co = i >> 6, ci = i & 63;
  o[i] = (_Float16)tw[co * 192 + ci * 3 + 1];
}

// ============================================================================
// Batched f16 WMMA GEMM: C = A(MxK, row-major) x B(KxN, row-major), f32 accum.
// Block = 256 threads (8 waves), tile 128x64; wave tile 32x32 (4 wmma / K-step,
// each A/B fragment reused twice -> 4 dword A + 8 f16 B loads per WMMA).
// Fragment layouts per CDNA5 ISA 7.12.2:
//   A 16x32: lane = M (0-15) | K-half (lane>>4); vgpr g holds K pair
//            base(g) = {0,2,4,6,16,18,20,22}[g] + 8*(lane>>4)
//   B 32x16: lane = N (0-15) | K-half (lane>>4); half s holds K in A's order
//   C 16x16: lane = N (0-15) | M-half; vgpr r holds M = r + 8*(lane>>4)
// Epilogue: mode 1 = per-channel scale/shift (folded BN) + ReLU -> f16
//           mode 2 = transposed f32 store + bias (final conv_joint)
// Dims divide the tiles (or waves exit early, wave-uniform), so EXEC stays
// all-ones inside every wave that executes WMMA.
// ============================================================================
__global__ __launch_bounds__(256) void k_gemm_f16(
    const _Float16* __restrict__ A, const _Float16* __restrict__ B,
    int M, int N, int K, int lda, int ldb,
    long long sA, long long sB, long long sOut,
    int mode,
    _Float16* __restrict__ outH, int ldoH,
    float* __restrict__ outF, int ldoF,
    const float* __restrict__ scale, const float* __restrict__ shift, int rowMod,
    const float* __restrict__ bias) {
  int bat = blockIdx.z;
  A += (size_t)bat * sA;
  B += (size_t)bat * sB;
  int lane = threadIdx.x & 31;
  int wave = threadIdx.x >> 5;
  int wm = wave & 3, wn = wave >> 2;
  int rowBase = blockIdx.y * 128 + wm * 32;
  int colBase = blockIdx.x * 64 + wn * 32;
  if (rowBase >= M || colBase >= N) return;   // wave-uniform

  int am = lane & 15, khalf = lane >> 4;
  const _Float16* Arow0 = A + (size_t)(rowBase + am) * lda;
  const _Float16* Arow1 = Arow0 + (size_t)16 * lda;
  const _Float16* Bcol = B + colBase;

  v8f acc00 = {0.f, 0.f, 0.f, 0.f, 0.f, 0.f, 0.f, 0.f};
  v8f acc01 = acc00, acc10 = acc00, acc11 = acc00;

  union AFrag { unsigned u[8]; v16h v; };
  union BFrag { uint4 q[2]; v16h v; };

  for (int k0 = 0; k0 < K; k0 += 32) {
    AFrag a0, a1;
    #pragma unroll
    for (int g = 0; g < 8; ++g) {
      int base = ((g < 4) ? (2 * g) : (16 + 2 * (g - 4))) + 8 * khalf;
      a0.u[g] = *(const unsigned*)(Arow0 + k0 + base);   // packed K, K+1
      a1.u[g] = *(const unsigned*)(Arow1 + k0 + base);
    }
    const _Float16* Bk = Bcol + (size_t)k0 * ldb + (lane & 15);
    BFrag b0, b1;
    #pragma unroll
    for (int s = 0; s < 16; ++s) {
      int kk = ((s < 8) ? s : (8 + s)) + 8 * khalf;
      b0.v[s] = Bk[(size_t)kk * ldb];
      b1.v[s] = Bk[(size_t)kk * ldb + 16];
    }
    acc00 = __builtin_amdgcn_wmma_f32_16x16x32_f16(false, a0.v, false, b0.v,
                                                   (short)0, acc00, false, false);
    acc01 = __builtin_amdgcn_wmma_f32_16x16x32_f16(false, a0.v, false, b1.v,
                                                   (short)0, acc01, false, false);
    acc10 = __builtin_amdgcn_wmma_f32_16x16x32_f16(false, a1.v, false, b0.v,
                                                   (short)0, acc10, false, false);
    acc11 = __builtin_amdgcn_wmma_f32_16x16x32_f16(false, a1.v, false, b1.v,
                                                   (short)0, acc11, false, false);
  }

  int cn = lane & 15, mh = lane >> 4;
  #pragma unroll
  for (int ms = 0; ms < 2; ++ms) {
    #pragma unroll
    for (int r = 0; r < 8; ++r) {
      int gm = rowBase + ms * 16 + r + 8 * mh;
      int gn0 = colBase + cn, gn1 = colBase + 16 + cn;
      float v0 = ms ? acc10[r] : acc00[r];
      float v1 = ms ? acc11[r] : acc01[r];
      if (mode == 1) {
        float sc = scale[gm % rowMod], sf = shift[gm % rowMod];
        _Float16* o = outH + (size_t)bat * sOut + (size_t)gm * ldoH;
        o[gn0] = (_Float16)fmaxf(v0 * sc + sf, 0.f);
        o[gn1] = (_Float16)fmaxf(v1 * sc + sf, 0.f);
      } else {
        float bb = bias ? bias[gm] : 0.f;
        outF[(size_t)gn0 * ldoF + gm] = v0 + bb;
        outF[(size_t)gn1 * ldoF + gm] = v1 + bb;
      }
    }
  }
}

// ============================================================================
static inline unsigned cdiv(long long a, long long b) { return (unsigned)((a + b - 1) / b); }

extern "C" void kernel_launch(void* const* d_in, const int* in_sizes, int n_in,
                              void* d_out, int out_size, void* d_ws, size_t ws_size,
                              hipStream_t stream) {
  auto in = [&](int i) { return (const float*)d_in[i]; };

  // ---- workspace carve-up (all 256B aligned; ~113 MB total) ----
  char* w = (char*)d_ws;
  size_t off = 0;
  auto alloc = [&](size_t bytes) -> char* {
    char* p = w + off;
    off += (bytes + 255) & ~(size_t)255;
    return p;
  };
  float*    prep = (float*)   alloc(512 * sizeof(float));
  float*    f    = (float*)   alloc((size_t)128 * 1024 * 2 * 4);        // feat+pos
  _Float16* HC1  = (_Float16*)alloc((size_t)8192 * 3072 * 2);           // block1 gcn out
  _Float16* A1h  = (_Float16*)alloc((size_t)3072 * 1024 * 2);           // A*imp1 f16
  _Float16* A2h  = (_Float16*)alloc((size_t)3072 * 1024 * 2);           // A*imp2 f16
  _Float16* Y1   = (_Float16*)alloc((size_t)8192 * 1024 * 2);           // bn1+relu
  _Float16* W1h  = (_Float16*)alloc((size_t)64 * 64 * 2);               // tcn1 center
  _Float16* X1   = (_Float16*)alloc((size_t)8192 * 1024 * 2);           // block1 out
  _Float16* HC2  = (_Float16*)alloc((size_t)256 * 3072 * 2);            // block2 gcn out
  _Float16* Y2   = (_Float16*)alloc((size_t)256 * 1024 * 2);            // bn1+relu
  _Float16* X2T  = (_Float16*)alloc((size_t)2048 * 128 * 2);            // block2 out^T
  _Float16* WJh  = (_Float16*)alloc((size_t)3072 * 2048 * 2);           // conv_joint w f16
  (void)ws_size;

  // ---- constants / folded BN params ----
  PrepArgs pa;
  pa.sk_g = in(SK_BNG); pa.sk_b = in(SK_BNB); pa.sk_m = in(SK_BNM); pa.sk_v = in(SK_BNV);
  pa.b1g = in(B1_BN1G); pa.b1b = in(B1_BN1B); pa.b1m = in(B1_BN1M); pa.b1v = in(B1_BN1V);
  pa.b2g = in(B1_BN2G); pa.b2b = in(B1_BN2B); pa.b2m = in(B1_BN2M); pa.b2v = in(B1_BN2V);
  pa.tcnb1 = in(B1_TCNB);
  pa.k2g = in(B2_BN1G); pa.k2b = in(B2_BN1B); pa.k2m = in(B2_BN1M); pa.k2v = in(B2_BN1V);
  pa.pe_b1 = in(PE_B1); pa.pe_g = in(PE_BNG); pa.pe_b = in(PE_BNB);
  pa.pe_m = in(PE_BNM); pa.pe_v = in(PE_BNV); pa.pe_w2 = in(PE_W2); pa.pe_b2 = in(PE_B2);
  k_prep<<<1, 64, 0, stream>>>(pa, prep);

  // ---- weight conversions ----
  {
    long long nA = (long long)3 * 1024 * 1024;
    k_cvt<<<cdiv(nA, 256), 256, 0, stream>>>(in(IN_A), in(B1_IMP), A1h, nA);
    k_cvt<<<cdiv(nA, 256), 256, 0, stream>>>(in(IN_A), in(B2_IMP), A2h, nA);
    long long nW = (long long)3072 * 2048;
    k_cvt<<<cdiv(nW, 256), 256, 0, stream>>>(in(CJ_W), nullptr, WJh, nW);
    k_w1<<<cdiv(64 * 64, 256), 256, 0, stream>>>(in(B1_TCNW), W1h);
  }

  // ---- features + block1 gcn expansion ----
  k_feat<<<cdiv(128 * 1024, 256), 256, 0, stream>>>(in(IN_JOINTS), in(SK_W1), in(SK_B1),
                                                    in(SK_W2), in(SK_B2), prep, f);
  {
    long long n = (long long)8192 * 3072;
    k_gcn1<<<cdiv(n, 256), 256, 0, stream>>>(f, in(B1_GCNW), in(B1_GCNB), HC1);
  }

  // ---- GEMM1: [8192x3072] x [3072x1024] -> bn1+relu -> Y1 (f16) ----
  k_gemm_f16<<<dim3(1024 / 64, 8192 / 128, 1), 256, 0, stream>>>(
      HC1, A1h, 8192, 1024, 3072, 3072, 1024, 0, 0, 0,
      1, Y1, 1024, nullptr, 0, prep + 128, prep + 192, 64, nullptr);

  // ---- tcn1 as batched GEMM: W1[64x64] x Y1_b[64x1024] -> bn2(+bias)+relu -> X1 ----
  k_gemm_f16<<<dim3(1024 / 64, 1, 128), 256, 0, stream>>>(
      W1h, Y1, 64, 1024, 64, 64, 1024, 0, (long long)64 * 1024, (long long)64 * 1024,
      1, X1, 1024, nullptr, 0, prep + 256, prep + 320, 64, nullptr);

  // ---- block2 gcn expansion ----
  k_gcn2<<<cdiv(128 * 6 * 1024, 256), 256, 0, stream>>>(X1, in(B2_GCNW), in(B2_GCNB), HC2);

  // ---- GEMM2: [256x3072] x [3072x1024] -> bn1+relu -> Y2 (f16) ----
  k_gemm_f16<<<dim3(1024 / 64, 256 / 128, 1), 256, 0, stream>>>(
      HC2, A2h, 256, 1024, 3072, 3072, 1024, 0, 0, 0,
      1, Y2, 1024, nullptr, 0, prep + 384, prep + 386, 2, nullptr);

  // ---- block2 tcn + bn2 + residual + relu, transposed store ----
  k_tcn2res<<<cdiv(128 * 2 * 1024, 256), 256, 0, stream>>>(
      Y2, X1, in(B2_TCNW), in(B2_TCNB),
      in(B2_BN2G), in(B2_BN2B), in(B2_BN2M), in(B2_BN2V),
      in(B2_RESW), in(B2_RESB), in(B2_RBG), in(B2_RBB), in(B2_RBM), in(B2_RBV), X2T);

  // ---- GEMM3 (conv_joint): Wj[3072x2048] x X2T[2048x128] -> d_out (transposed + bias) ----
  k_gemm_f16<<<dim3(128 / 64, 3072 / 128, 1), 256, 0, stream>>>(
      WJh, X2T, 3072, 128, 2048, 2048, 128, 0, 0, 0,
      2, nullptr, 0, (float*)d_out, 3072, nullptr, nullptr, 1, in(CJ_B));
}
